// AtencaoMultihead_52948356825375
// MI455X (gfx1250) — matrix operand order repaired and checked by reading, not compiled, not run on hardware
//
#include <hip/hip_runtime.h>

// ---------------------------------------------------------------------------
// Multi-head attention forward, MI455X (gfx1250, wave32, WMMA bf16).
//   B=2, S=2048, D=1024, H=16, DK=64
// Pipeline:
//   0) cvt_w     : Wq/Wk/Wv/Wo f32 -> bf16 (once, removes cvts from GEMM loops)
//   1) gemm_proj : Q = q@Wq^T+bq -> bf16 [B,H,S,DK]
//                  K = k@Wk^T+bk -> bf16 [B,H,S,DK]
//                  V = v@Wv^T+bv -> bf16 [B,H,DK,S] (transposed for PV NT gemm)
//   2) attn      : flash-attention, causal, online softmax, WMMA bf16
//   3) gemm_out  : out = ctx@Wo^T+bo -> f32 [B,S,D]
// GEMM waves compute 32x64 tiles, k-loop unrolled x2 with ping-pong fragment
// sets (no rotation movs, no WMMA->VALU WAR hazard NOPs).
// ---------------------------------------------------------------------------

typedef __bf16 bf16;
typedef __attribute__((ext_vector_type(16))) __bf16 v16bf;
typedef __attribute__((ext_vector_type(8)))  float  v8f;

#define BATCH 2
#define S_LEN 2048
#define D_DIM 1024
#define NH    16
#define DKH   64

__device__ inline v8f wmma_bf16(v16bf a, v16bf b, v8f c) {
  return __builtin_amdgcn_wmma_f32_16x16x32_bf16(
      false, a, false, b, (short)0, c, false, false);
}

// A-fragment (16x32, M = lane&15): elems 0..15 = row[k0+2*kb .. +15],
// kb = 8*(lane>>4).  Two 16B loads, contiguous 32B.
__device__ inline v16bf load_a_bf16(const bf16* row, int k0, int kb) {
  union { uint4 u[2]; v16bf v; } r;
  r.u[0] = *(const uint4*)(row + k0 + 2 * kb);
  r.u[1] = *(const uint4*)(row + k0 + 2 * kb + 8);
  return r.v;
}

// B-fragment (32x16, N = lane&15): elems 0..15 = row[k0 .. +15]
// (caller folds in kb16 = 16*(lane>>4)).  Two 16B loads, contiguous 32B.
__device__ inline v16bf load_b_bf16(const bf16* row, int k0) {
  union { uint4 u[2]; v16bf v; } r;
  r.u[0] = *(const uint4*)(row + k0);
  r.u[1] = *(const uint4*)(row + k0 + 8);
  return r.v;
}

// Convert two 8-float runs into one bf16 A-fragment (for f32 activations).
__device__ inline v16bf cvt8x2(const float* p0, const float* p1) {
  v16bf v;
#pragma unroll
  for (int i = 0; i < 8; ++i) {
    v[i]     = (bf16)p0[i];
    v[8 + i] = (bf16)p1[i];
  }
  return v;
}

// Fragment set for one 32x64-tile k-step.
struct FragSet { v16bf a0, a1, b0, b1, b2, b3; };

__device__ inline void mma8(v8f acc[8], const FragSet& f) {
  acc[0] = wmma_bf16(f.a0, f.b0, acc[0]);
  acc[1] = wmma_bf16(f.a0, f.b1, acc[1]);
  acc[2] = wmma_bf16(f.a0, f.b2, acc[2]);
  acc[3] = wmma_bf16(f.a0, f.b3, acc[3]);
  acc[4] = wmma_bf16(f.a1, f.b0, acc[4]);
  acc[5] = wmma_bf16(f.a1, f.b1, acc[5]);
  acc[6] = wmma_bf16(f.a1, f.b2, acc[6]);
  acc[7] = wmma_bf16(f.a1, f.b3, acc[7]);
}

// ---------------------------------------------------------------------------
// Stage 0: bulk f32 -> bf16 (weights).  8 elems/thread, vectorized.
// ---------------------------------------------------------------------------
__global__ void __launch_bounds__(256)
cvt_w_kernel(const float* __restrict__ in, bf16* __restrict__ out, int n) {
  int i = (blockIdx.x * blockDim.x + threadIdx.x) * 8;
  if (i >= n) return;
  const float4 x0 = *(const float4*)(in + i);
  const float4 x1 = *(const float4*)(in + i + 4);
  union { bf16 e[8]; uint4 u; } r;
  r.e[0] = (bf16)x0.x; r.e[1] = (bf16)x0.y; r.e[2] = (bf16)x0.z; r.e[3] = (bf16)x0.w;
  r.e[4] = (bf16)x1.x; r.e[5] = (bf16)x1.y; r.e[6] = (bf16)x1.z; r.e[7] = (bf16)x1.w;
  *(uint4*)(out + i) = r.u;
}

// ---------------------------------------------------------------------------
// Stage 1: C = A(f32)[4096x1024] @ Wb(bf16)^T + bias -> bf16 (scatter by mode)
// One wave = 32x64 tile: 8 WMMAs / k-step, ping-pong double buffering.
// ---------------------------------------------------------------------------
__global__ void __launch_bounds__(128)
gemm_proj_kernel(const float* __restrict__ A, const bf16* __restrict__ Wb,
                 const float* __restrict__ bias, bf16* __restrict__ out,
                 int mode) {
  const int lane  = threadIdx.x & 31;
  const int wave  = blockIdx.x * (blockDim.x >> 5) + (threadIdx.x >> 5);
  const int NT    = D_DIM / 64;                 // 16 n-tiles
  const int mt    = wave / NT, nt = wave % NT;
  const int n0    = mt * 32, j0 = nt * 64;
  const int mrow  = lane & 15, khalf = lane >> 4;
  const int kbA   = 8 * khalf, kbB = 16 * khalf;

  const float* arow0 = A + (size_t)(n0 + mrow) * D_DIM;
  const float* arow1 = arow0 + (size_t)16 * D_DIM;
  const bf16*  wrow[4];
#pragma unroll
  for (int t = 0; t < 4; ++t)
    wrow[t] = Wb + (size_t)(j0 + t * 16 + mrow) * D_DIM;

  auto loadf = [&](FragSet& f, int k0) {
    f.a0 = cvt8x2(arow0 + k0 + 2 * kbA, arow0 + k0 + 2 * kbA + 8);
    f.a1 = cvt8x2(arow1 + k0 + 2 * kbA, arow1 + k0 + 2 * kbA + 8);
    f.b0 = load_b_bf16(wrow[0], k0 + kbB);
    f.b1 = load_b_bf16(wrow[1], k0 + kbB);
    f.b2 = load_b_bf16(wrow[2], k0 + kbB);
    f.b3 = load_b_bf16(wrow[3], k0 + kbB);
  };

  v8f acc[8] = {};
  FragSet f0, f1;
  loadf(f0, 0);
  // D/32 = 32 k-steps: 15 pipelined double-iterations + peeled tail pair.
  for (int k0 = 0; k0 < D_DIM - 64; k0 += 64) {
    loadf(f1, k0 + 32);
    mma8(acc, f0);
    loadf(f0, k0 + 64);
    mma8(acc, f1);
  }
  loadf(f1, D_DIM - 32);
  mma8(acc, f0);
  mma8(acc, f1);

#pragma unroll
  for (int half = 0; half < 2; ++half) {
#pragma unroll
    for (int t = 0; t < 4; ++t) {
      const int j  = j0 + t * 16 + mrow;        // output column (C-frag N)
      const float bj = bias[j];
      const int h = j >> 6, dk = j & 63;
#pragma unroll
      for (int i = 0; i < 8; ++i) {
        const int n = n0 + half * 16 + i + 8 * khalf;  // output row
        const int b_ = n / S_LEN, s = n % S_LEN;
        const float v = acc[half * 4 + t][i] + bj;
        size_t off = (mode == 0)
            ? ((size_t)((b_ * NH + h) * S_LEN + s)) * DKH + dk   // [B,H,S,DK]
            : ((size_t)((b_ * NH + h) * DKH + dk)) * S_LEN + s;  // [B,H,DK,S]
        out[off] = (bf16)v;
      }
    }
  }
}

// ---------------------------------------------------------------------------
// Stage 2: flash attention.  One wave = one (b,h) x 16-query tile.
// Streams keys in chunks of 32; online softmax in f32; P -> LDS -> A-frag.
// ---------------------------------------------------------------------------
__global__ void __launch_bounds__(128)
attn_kernel(const bf16* __restrict__ qh, const bf16* __restrict__ kh,
            const bf16* __restrict__ vT, bf16* __restrict__ ctx) {
  __shared__ alignas(16) bf16 Pb[4][16 * 32];   // per-wave P staging (1KB each)

  const int lane  = threadIdx.x & 31;
  const int wslot = threadIdx.x >> 5;
  const int wave  = blockIdx.x * 4 + wslot;
  const int QT    = S_LEN / 16;                 // 128 query tiles per (b,h)
  const int bh    = wave / QT;
  const int q0    = (wave % QT) * 16;
  const int mrow  = lane & 15, khalf = lane >> 4;
  const float scale = 0.125f;                   // 1/sqrt(DK)

  // Q A-fragments for DK=64 (two K-steps of 32), loaded once.
  const bf16* qrow = qh + ((size_t)bh * S_LEN + q0 + mrow) * DKH;
  const v16bf aQ0 = load_a_bf16(qrow, 0,  8 * khalf);
  const v16bf aQ1 = load_a_bf16(qrow, 32, 8 * khalf);

  const bf16* vrow[4];
#pragma unroll
  for (int t = 0; t < 4; ++t)
    vrow[t] = vT + ((size_t)bh * DKH + t * 16 + mrow) * S_LEN;

  float m[8], l[8];
  v8f acc[4] = {};
#pragma unroll
  for (int i = 0; i < 8; ++i) { m[i] = -1e30f; l[i] = 0.f; }

  bf16* Pw = Pb[wslot];
  const int kend = q0 + 16;                     // causal: keys < q0+16

  for (int k0 = 0; k0 < kend; k0 += 32) {
    // ---- scores S = Q K^T (16 queries x 32 keys), 4 WMMAs -----------------
    const bf16* krow0 = kh + ((size_t)bh * S_LEN + k0 + mrow) * DKH;
    const bf16* krow1 = krow0 + 16 * DKH;
    v8f s0 = {}, s1 = {};
    s0 = wmma_bf16(aQ0, load_b_bf16(krow0, 16 * khalf),      s0);
    s0 = wmma_bf16(aQ1, load_b_bf16(krow0, 32 + 16 * khalf), s0);
    s1 = wmma_bf16(aQ0, load_b_bf16(krow1, 16 * khalf),      s1);
    s1 = wmma_bf16(aQ1, load_b_bf16(krow1, 32 + 16 * khalf), s1);

    // V B-fragments are independent of softmax: issue their loads now so the
    // exp/shuffle VALU chain below hides the latency.
    v16bf bV0 = load_b_bf16(vrow[0], k0 + 16 * khalf);
    v16bf bV1 = load_b_bf16(vrow[1], k0 + 16 * khalf);
    v16bf bV2 = load_b_bf16(vrow[2], k0 + 16 * khalf);
    v16bf bV3 = load_b_bf16(vrow[3], k0 + 16 * khalf);

    const bool diag = (k0 + 31 >= q0);          // chunk may cross diagonal
#pragma unroll
    for (int i = 0; i < 8; ++i) {
      float v0 = s0[i] * scale, v1 = s1[i] * scale;
      if (diag) {
        const int rg = q0 + i + 8 * khalf;      // global query row
        if (k0 + mrow      > rg) v0 = -1e30f;   // col = k0 + (lane&15)
        if (k0 + 16 + mrow > rg) v1 = -1e30f;   // col = k0+16+(lane&15)
      }
      float mx = fmaxf(v0, v1);
#pragma unroll
      for (int off = 8; off; off >>= 1) mx = fmaxf(mx, __shfl_xor(mx, off, 16));
      const float mn    = fmaxf(m[i], mx);
      const float alpha = __expf(m[i] - mn);
      const float p0 = __expf(v0 - mn), p1 = __expf(v1 - mn);
      float sum = p0 + p1;
#pragma unroll
      for (int off = 8; off; off >>= 1) sum += __shfl_xor(sum, off, 16);
      m[i] = mn;
      l[i] = l[i] * alpha + sum;
#pragma unroll
      for (int t = 0; t < 4; ++t) acc[t][i] *= alpha;
      const int row = i + 8 * khalf;
      Pw[row * 32 + mrow]      = (bf16)p0;
      Pw[row * 32 + 16 + mrow] = (bf16)p1;
    }

    // ---- reload P as A-fragment (same-wave DS ops are in order) -----------
    union { uint4 u[2]; v16bf v; } pa;
    pa.u[0] = *(const uint4*)(&Pw[mrow * 32 + 16 * khalf]);
    pa.u[1] = *(const uint4*)(&Pw[mrow * 32 + 16 * khalf + 8]);

    // ---- O += P V  (K=32 keys, 4 dk-tiles of 16), 4 WMMAs -----------------
    acc[0] = wmma_bf16(pa.v, bV0, acc[0]);
    acc[1] = wmma_bf16(pa.v, bV1, acc[1]);
    acc[2] = wmma_bf16(pa.v, bV2, acc[2]);
    acc[3] = wmma_bf16(pa.v, bV3, acc[3]);
  }

  // epilogue: normalize and write heads-merged context [B,S,D] in bf16
  const int b_ = bh / NH, h = bh % NH;
#pragma unroll
  for (int t = 0; t < 4; ++t) {
#pragma unroll
    for (int i = 0; i < 8; ++i) {
      const int row = q0 + i + 8 * khalf;
      const float v = acc[t][i] / l[i];
      size_t off = ((size_t)(b_ * S_LEN + row)) * D_DIM + h * DKH + t * 16 + mrow;
      ctx[off] = (bf16)v;
    }
  }
}

// ---------------------------------------------------------------------------
// Stage 3: out(f32) = ctx(bf16)[4096x1024] @ Wo(bf16)^T + bo.  32x64 tiles.
// ---------------------------------------------------------------------------
__global__ void __launch_bounds__(128)
gemm_out_kernel(const bf16* __restrict__ A, const bf16* __restrict__ Wb,
                const float* __restrict__ bias, float* __restrict__ out) {
  const int lane  = threadIdx.x & 31;
  const int wave  = blockIdx.x * (blockDim.x >> 5) + (threadIdx.x >> 5);
  const int NT    = D_DIM / 64;
  const int mt    = wave / NT, nt = wave % NT;
  const int n0    = mt * 32, j0 = nt * 64;
  const int mrow  = lane & 15, khalf = lane >> 4;
  const int kbA   = 8 * khalf, kbB = 16 * khalf;

  const bf16* arow0 = A + (size_t)(n0 + mrow) * D_DIM;
  const bf16* arow1 = arow0 + (size_t)16 * D_DIM;
  const bf16* wrow[4];
#pragma unroll
  for (int t = 0; t < 4; ++t)
    wrow[t] = Wb + (size_t)(j0 + t * 16 + mrow) * D_DIM;

  auto loadf = [&](FragSet& f, int k0) {
    f.a0 = load_a_bf16(arow0, k0, kbA);
    f.a1 = load_a_bf16(arow1, k0, kbA);
    f.b0 = load_b_bf16(wrow[0], k0 + kbB);
    f.b1 = load_b_bf16(wrow[1], k0 + kbB);
    f.b2 = load_b_bf16(wrow[2], k0 + kbB);
    f.b3 = load_b_bf16(wrow[3], k0 + kbB);
  };

  v8f acc[8] = {};
  FragSet f0, f1;
  loadf(f0, 0);
  for (int k0 = 0; k0 < D_DIM - 64; k0 += 64) {
    loadf(f1, k0 + 32);
    mma8(acc, f0);
    loadf(f0, k0 + 64);
    mma8(acc, f1);
  }
  loadf(f1, D_DIM - 32);
  mma8(acc, f0);
  mma8(acc, f1);

#pragma unroll
  for (int half = 0; half < 2; ++half) {
#pragma unroll
    for (int t = 0; t < 4; ++t) {
      const int j = j0 + t * 16 + mrow;
      const float bj = bias[j];
#pragma unroll
      for (int i = 0; i < 8; ++i) {
        const int n = n0 + half * 16 + i + 8 * khalf;
        out[(size_t)n * D_DIM + j] = acc[half * 4 + t][i] + bj;
      }
    }
  }
}

// ---------------------------------------------------------------------------
extern "C" void kernel_launch(void* const* d_in, const int* in_sizes, int n_in,
                              void* d_out, int out_size, void* d_ws,
                              size_t ws_size, hipStream_t stream) {
  (void)in_sizes; (void)n_in; (void)out_size; (void)ws_size;

  const float* q  = (const float*)d_in[0];
  const float* k  = (const float*)d_in[1];
  const float* v  = (const float*)d_in[2];
  // d_in[3] = mask (int32 tril) — causality applied analytically in-kernel.
  const float* Wq = (const float*)d_in[4];
  const float* bq = (const float*)d_in[5];
  const float* Wk = (const float*)d_in[6];
  const float* bk = (const float*)d_in[7];
  const float* Wv = (const float*)d_in[8];
  const float* bv = (const float*)d_in[9];
  const float* Wo = (const float*)d_in[10];
  const float* bo = (const float*)d_in[11];

  const size_t headElems = (size_t)BATCH * NH * S_LEN * DKH;  // 4,194,304
  const size_t wElems    = (size_t)D_DIM * D_DIM;             // 1,048,576
  bf16* qh  = (bf16*)d_ws;             // 8 MB  [B,H,S,DK]
  bf16* kh  = qh + headElems;          // 8 MB  [B,H,S,DK]
  bf16* vT  = kh + headElems;          // 8 MB  [B,H,DK,S]
  bf16* ctx = vT + headElems;          // 8 MB  [B,S,D]
  bf16* wqb = ctx + headElems;         // 2 MB
  bf16* wkb = wqb + wElems;            // 2 MB
  bf16* wvb = wkb + wElems;            // 2 MB
  bf16* wob = wvb + wElems;            // 2 MB   (total ws: 40 MB)

  const dim3 blk(128);
  const dim3 cvt_grid((unsigned)(wElems / 8 / 256));          // 512 blocks
  const int gemm_waves = (BATCH * S_LEN / 32) * (D_DIM / 64); // 2048
  const dim3 gemm_grid(gemm_waves / 4);                       // 512 blocks
  const int attn_waves = BATCH * NH * (S_LEN / 16);           // 4096
  const dim3 attn_grid(attn_waves / 4);                       // 1024 blocks

  cvt_w_kernel<<<cvt_grid, 256, 0, stream>>>(Wq, wqb, (int)wElems);
  cvt_w_kernel<<<cvt_grid, 256, 0, stream>>>(Wk, wkb, (int)wElems);
  cvt_w_kernel<<<cvt_grid, 256, 0, stream>>>(Wv, wvb, (int)wElems);
  cvt_w_kernel<<<cvt_grid, 256, 0, stream>>>(Wo, wob, (int)wElems);

  gemm_proj_kernel<<<gemm_grid, blk, 0, stream>>>(q, wqb, bq, qh, 0);
  gemm_proj_kernel<<<gemm_grid, blk, 0, stream>>>(k, wkb, bk, kh, 0);
  gemm_proj_kernel<<<gemm_grid, blk, 0, stream>>>(v, wvb, bv, vT, 1);
  attn_kernel<<<attn_grid, blk, 0, stream>>>(qh, kh, vT, ctx);
  gemm_out_kernel<<<gemm_grid, blk, 0, stream>>>(ctx, wob, bo, (float*)d_out);
}
